// Encoder_85289460564606
// MI455X (gfx1250) — matrix operand order repaired and checked by reading, not compiled, not run on hardware
//
#include <hip/hip_runtime.h>
#include <hip/hip_bf16.h>
#include <stdint.h>

// ---------------------------------------------------------------------------
// Stacked LSTM encoder for MI455X (gfx1250, wave32).
//   layer1: bidirectional LSTM (forward + backward cells), D=512
//   layer2: LSTM D=1024, layer3/4: LSTM D=512
// Strategy:
//   * f16 WMMA (v_wmma_f32_16x16x32_f16) for z = [x_t ; h_{t-1}] @ [W;U]^T + b
//   * weights pre-transposed to f16 [4U, D+512] so B-frags are contiguous
//   * persistent cooperative kernel per cell: 16 WGs x 256 thr, grid barrier
//     per time step (serial dependence over T=512), h double-buffered f16,
//     c kept in registers, z staged in LDS for the gate math.
// ---------------------------------------------------------------------------

typedef _Float16 half_t;
typedef __attribute__((ext_vector_type(16))) _Float16 v16h;
typedef __attribute__((ext_vector_type(8)))  _Float16 v8h;
typedef __attribute__((ext_vector_type(8)))  float    v8f;

#define UNITS 512
#define BATCH 64
#define SEQT  512
#define NBLK  16          // cooperative grid size for the recurrent kernel
#define COLS_PER_WG 32    // 512 / NBLK columns of each gate per WG

union V16 { v16h v; v8h h[2]; };

__device__ __forceinline__ float sigmoidf_(float x) {
  return 1.0f / (1.0f + __expf(-x));
}

// ---------------------------------------------------------------------------
// Embedding gather, fp32 -> f16, stored t-major [T, B, 512]
// ---------------------------------------------------------------------------
__global__ void embed_kernel(const int* __restrict__ tok,
                             const float* __restrict__ emb,
                             half_t* __restrict__ Xe) {
  int t = blockIdx.x;   // 0..T-1
  int b = blockIdx.y;   // 0..B-1
  int tk = tok[(size_t)b * SEQT + t];
  const float* src = emb + (size_t)tk * 512;
  half_t* dst = Xe + ((size_t)t * BATCH + b) * 512;
  for (int e = threadIdx.x; e < 512; e += blockDim.x)
    dst[e] = (half_t)src[e];
}

// ---------------------------------------------------------------------------
// Build f16 transposed concatenated weights: Wt[n][k] = (k<DIN ? W[k][n] : U[k-DIN][n])
// Wt is [4*UNITS, DIN+UNITS] row-major (rows = output column n of z).
// ---------------------------------------------------------------------------
__global__ void prep_w(const float* __restrict__ W,
                       const float* __restrict__ U,
                       half_t* __restrict__ Wt, int DIN) {
  int ktot = DIN + UNITS;
  size_t idx = (size_t)blockIdx.x * blockDim.x + threadIdx.x;
  size_t total = (size_t)4 * UNITS * ktot;
  if (idx >= total) return;
  int n = (int)(idx / ktot);
  int k = (int)(idx % ktot);
  float v = (k < DIN) ? W[(size_t)k * (4 * UNITS) + n]
                      : U[(size_t)(k - DIN) * (4 * UNITS) + n];
  Wt[idx] = (half_t)v;
}

// ---------------------------------------------------------------------------
// Grid barrier (two-phase, generation counter). 16 WGs are trivially
// co-resident on MI455X.
// ---------------------------------------------------------------------------
struct GridBar { unsigned cnt; unsigned gen; unsigned pad[14]; };

__device__ __forceinline__ void grid_barrier(GridBar* bar) {
  __syncthreads();
  if (threadIdx.x == 0) {
    __threadfence();                       // release h stores to device scope
    volatile unsigned* genp = &bar->gen;
    unsigned g = *genp;
    if (atomicAdd(&bar->cnt, 1u) == NBLK - 1) {
      bar->cnt = 0;
      __threadfence();
      atomicAdd(&bar->gen, 1u);
    } else {
      while (*genp == g) { __builtin_amdgcn_s_sleep(1); }
    }
  }
  __syncthreads();
  __threadfence();                         // acquire: refetch h written by peers
}

// ---------------------------------------------------------------------------
// Persistent recurrent LSTM cell.
//   X    : [T, B, DIN] f16 (t-major)
//   Wt   : [4U, DIN+U] f16 (pre-transposed)
//   bias : [4U] f32
//   hbuf : [2, B, U] f16 double buffer (zeroed before launch)
//   Yseq : [T, B, ldy] f16 seq output at column offset ycol0 (or null)
//   outF : [B, T, U] f32 seq output (layer 4 -> d_out, or null)
//   hcF  : f32, h_T at [0, B*U), c_T at [B*U, 2*B*U) (or null)
// Per WG: 32 columns of each of the 4 gates. Per wave (8 waves):
//   rowtile rt = w&3 (16 rows), gate pair p = w>>2 -> 4 accum tiles.
// ---------------------------------------------------------------------------
__global__ void __launch_bounds__(256, 1)
lstm_cell_kernel(const half_t* __restrict__ X,
                 const half_t* __restrict__ Wt,
                 const float*  __restrict__ bias,
                 half_t* __restrict__ hbuf,
                 half_t* __restrict__ Yseq,
                 float*  __restrict__ outF,
                 float*  __restrict__ hcF,
                 GridBar* bar,
                 int DIN, int ldy, int ycol0, int reverse) {
  const int KTOT = DIN + UNITS;
  const int tid  = threadIdx.x;
  const int lane = tid & 31;
  const int w    = tid >> 5;        // wave 0..7
  const int rt   = w & 3;           // row tile (16 rows each, 64 rows total)
  const int p    = w >> 2;          // gate pair: 0 -> gates {0,1}, 1 -> {2,3}
  const int colBase = blockIdx.x * COLS_PER_WG;  // gate-column slice base
  const int m  = lane & 15;         // A row / B column within tile
  const int kh = lane >> 4;         // half-wave selector for K layout

  __shared__ float zlds[64 * 128];  // [row 0..63][gate*32 + col 0..31], 32 KB

  // cell state registers: thread -> rows grow0..grow0+7, column colBase+gcol
  const int gcol  = tid & 31;
  const int grow0 = (tid >> 5) * 8;
  float creg[8];
#pragma unroll
  for (int i = 0; i < 8; ++i) creg[i] = 0.0f;

  // per-tile bias (column depends only on lane)
  float btile[4];
#pragma unroll
  for (int gi = 0; gi < 2; ++gi)
#pragma unroll
    for (int sc = 0; sc < 2; ++sc)
      btile[gi * 2 + sc] =
          bias[(2 * p + gi) * UNITS + colBase + sc * 16 + m];

  for (int s_step = 0; s_step < SEQT; ++s_step) {
    const int t_in = reverse ? (SEQT - 1 - s_step) : s_step;
    const half_t* hread  = hbuf + (size_t)(s_step & 1) * BATCH * UNITS;
    half_t*       hwrite = hbuf + (size_t)((s_step + 1) & 1) * BATCH * UNITS;

    v8f acc[4];
#pragma unroll
    for (int i = 0; i < 4; ++i)
#pragma unroll
      for (int r = 0; r < 8; ++r) acc[i][r] = 0.0f;

    const half_t* xrow = X + ((size_t)t_in * BATCH + 16 * rt + m) * DIN;
    const half_t* hrow = hread + (size_t)(16 * rt + m) * UNITS;

    // prefetch next step's x row (global_prefetch_b8)
    {
      int t_nx = reverse ? (t_in - 1) : (t_in + 1);
      if (t_nx >= 0 && t_nx < SEQT)
        __builtin_prefetch(X + ((size_t)t_nx * BATCH + 16 * rt + m) * DIN, 0, 0);
    }

    for (int kb = 0; kb < KTOT; kb += 32) {
      // A fragment (16-bit A 16x32 layout): two contiguous 16B loads
      const half_t* arow = (kb < DIN) ? (xrow + kb) : (hrow + (kb - DIN));
      V16 a;
      a.h[0] = *(const v8h*)(arow + 8 * kh);
      a.h[1] = *(const v8h*)(arow + 16 + 8 * kh);

#pragma unroll
      for (int gi = 0; gi < 2; ++gi) {
#pragma unroll
        for (int sc = 0; sc < 2; ++sc) {
          // B fragment: row n of Wt, K = 8*kh + e and 16 + 8*kh + e (two 8-half loads)
          int zcol = (2 * p + gi) * UNITS + colBase + sc * 16 + m;
          const half_t* brow = Wt + (size_t)zcol * KTOT + kb + 8 * kh;
          V16 bf;
          bf.h[0] = *(const v8h*)(brow);
          bf.h[1] = *(const v8h*)(brow + 16);
          acc[gi * 2 + sc] = __builtin_amdgcn_wmma_f32_16x16x32_f16(
              false, a.v, false, bf.v, (short)0, acc[gi * 2 + sc],
              false, false);
        }
      }
    }

    // write z tiles (+bias) to LDS; D layout: reg r -> row r + 8*(lane>>4)
#pragma unroll
    for (int gi = 0; gi < 2; ++gi)
#pragma unroll
      for (int sc = 0; sc < 2; ++sc) {
        int lc = (2 * p + gi) * 32 + sc * 16 + m;
#pragma unroll
        for (int r = 0; r < 8; ++r) {
          int row = 16 * rt + r + 8 * kh;
          zlds[row * 128 + lc] = acc[gi * 2 + sc][r] + btile[gi * 2 + sc];
        }
      }
    __syncthreads();

    // gate math: i,f,g,o -> c,h
#pragma unroll
    for (int i = 0; i < 8; ++i) {
      int row = grow0 + i;
      float iv = zlds[row * 128 + gcol];
      float fv = zlds[row * 128 + 32 + gcol];
      float gv = zlds[row * 128 + 64 + gcol];
      float ov = zlds[row * 128 + 96 + gcol];
      float cn = sigmoidf_(fv) * creg[i] + sigmoidf_(iv) * tanhf(gv);
      float hv = sigmoidf_(ov) * tanhf(cn);
      creg[i] = cn;
      int col = colBase + gcol;
      hwrite[(size_t)row * UNITS + col] = (half_t)hv;
      if (Yseq)
        Yseq[((size_t)t_in * BATCH + row) * ldy + ycol0 + col] = (half_t)hv;
      if (outF)
        outF[((size_t)row * SEQT + t_in) * UNITS + col] = hv;
      if (hcF && s_step == SEQT - 1) {
        hcF[(size_t)row * UNITS + col] = hv;                       // h_T
        hcF[(size_t)BATCH * UNITS + (size_t)row * UNITS + col] = cn; // c_T
      }
    }

    grid_barrier(bar);   // publish h(t) before anyone starts step t+1
  }
}

// ---------------------------------------------------------------------------
// Host launcher
// ---------------------------------------------------------------------------
extern "C" void kernel_launch(void* const* d_in, const int* in_sizes, int n_in,
                              void* d_out, int out_size, void* d_ws,
                              size_t ws_size, hipStream_t stream) {
  (void)in_sizes; (void)n_in; (void)out_size; (void)ws_size;

  const int*   tok = (const int*)d_in[0];
  const float* emb = (const float*)d_in[2];
  const float* W1f = (const float*)d_in[3];
  const float* U1f = (const float*)d_in[4];
  const float* b1f = (const float*)d_in[5];
  const float* W1b = (const float*)d_in[6];
  const float* U1b = (const float*)d_in[7];
  const float* b1b = (const float*)d_in[8];
  const float* W2  = (const float*)d_in[9];
  const float* U2  = (const float*)d_in[10];
  const float* b2  = (const float*)d_in[11];
  const float* W3  = (const float*)d_in[12];
  const float* U3  = (const float*)d_in[13];
  const float* b3  = (const float*)d_in[14];
  const float* W4  = (const float*)d_in[15];
  const float* U4  = (const float*)d_in[16];
  const float* b4  = (const float*)d_in[17];

  char* ws = (char*)d_ws;
  size_t off = 0;
  auto alloc = [&](size_t bytes) -> void* {
    void* pp = ws + off;
    off = (off + bytes + 255) & ~(size_t)255;
    return pp;
  };

  // sequence buffers (f16). Xe is reused as L2out, L1 region as L3out.
  half_t* Xe  = (half_t*)alloc((size_t)SEQT * BATCH * 512 * 2);   // 32 MB
  half_t* L1  = (half_t*)alloc((size_t)SEQT * BATCH * 1024 * 2);  // 64 MB
  half_t* Wt1f = (half_t*)alloc((size_t)2048 * 1024 * 2);
  half_t* Wt1b = (half_t*)alloc((size_t)2048 * 1024 * 2);
  half_t* Wt2  = (half_t*)alloc((size_t)2048 * 1536 * 2);
  half_t* Wt3  = (half_t*)alloc((size_t)2048 * 1024 * 2);
  half_t* Wt4  = (half_t*)alloc((size_t)2048 * 1024 * 2);
  half_t* hbuf = (half_t*)alloc((size_t)2 * BATCH * UNITS * 2);
  GridBar* bars = (GridBar*)alloc(8 * sizeof(GridBar));

  half_t* L2out = Xe;   // reuse: embeddings dead after layer-1 cells
  half_t* L3out = L1;   // reuse: layer-1 output dead after layer 2

  hipMemsetAsync(bars, 0, 8 * sizeof(GridBar), stream);

  // embeddings -> f16 [T, B, 512]
  embed_kernel<<<dim3(SEQT, BATCH), 128, 0, stream>>>(tok, emb, Xe);

  // weight prep (transpose + concat + f16)
  {
    int thr = 256;
    int n1024 = (int)(((size_t)2048 * 1024 + thr - 1) / thr);
    int n1536 = (int)(((size_t)2048 * 1536 + thr - 1) / thr);
    prep_w<<<n1024, thr, 0, stream>>>(W1f, U1f, Wt1f, 512);
    prep_w<<<n1024, thr, 0, stream>>>(W1b, U1b, Wt1b, 512);
    prep_w<<<n1536, thr, 0, stream>>>(W2,  U2,  Wt2,  1024);
    prep_w<<<n1024, thr, 0, stream>>>(W3,  U3,  Wt3,  512);
    prep_w<<<n1024, thr, 0, stream>>>(W4,  U4,  Wt4,  512);
  }

  const size_t hbytes = (size_t)2 * BATCH * UNITS * 2;
  float* outF = (float*)d_out;
  float* hcF  = (float*)d_out + (size_t)BATCH * SEQT * UNITS;

  // layer 1 forward: Xe -> L1[:, :, 0:512]
  hipMemsetAsync(hbuf, 0, hbytes, stream);
  lstm_cell_kernel<<<NBLK, 256, 0, stream>>>(
      Xe, Wt1f, b1f, hbuf, L1, nullptr, nullptr, &bars[0],
      512, 1024, 0, 0);

  // layer 1 backward: Xe (time-reversed) -> L1[:, :, 512:1024] (re-reversed)
  hipMemsetAsync(hbuf, 0, hbytes, stream);
  lstm_cell_kernel<<<NBLK, 256, 0, stream>>>(
      Xe, Wt1b, b1b, hbuf, L1, nullptr, nullptr, &bars[1],
      512, 1024, 512, 1);

  // layer 2: L1 (D=1024) -> L2out
  hipMemsetAsync(hbuf, 0, hbytes, stream);
  lstm_cell_kernel<<<NBLK, 256, 0, stream>>>(
      L1, Wt2, b2, hbuf, L2out, nullptr, nullptr, &bars[2],
      1024, 512, 0, 0);

  // layer 3: L2out -> L3out
  hipMemsetAsync(hbuf, 0, hbytes, stream);
  lstm_cell_kernel<<<NBLK, 256, 0, stream>>>(
      L2out, Wt3, b3, hbuf, L3out, nullptr, nullptr, &bars[3],
      512, 512, 0, 0);

  // layer 4: L3out -> d_out (f32 [B,T,512] + h_T + c_T)
  hipMemsetAsync(hbuf, 0, hbytes, stream);
  lstm_cell_kernel<<<NBLK, 256, 0, stream>>>(
      L3out, Wt4, b4, hbuf, nullptr, outF, hcF, &bars[4],
      512, 512, 0, 0);
}
